// MM_DecoderLSTM_16346645529091
// MI455X (gfx1250) — matrix operand order repaired and checked by reading, not compiled, not run on hardware
//
#include <hip/hip_runtime.h>
#include <hip/hip_bf16.h>

typedef _Float16 v16h __attribute__((ext_vector_type(16)));
typedef _Float16 v8h  __attribute__((ext_vector_type(8)));
typedef float    v8f  __attribute__((ext_vector_type(8)));

namespace {
constexpr int H_DIM   = 64;
constexpr int M_MODES = 6;
constexpr int T_STEPS = 30;
constexpr int WAVES   = 8;     // waves per block
constexpr int ROWS    = 16;    // agents per wave (WMMA M)
constexpr int BLK_B   = WAVES * ROWS;  // 128 agents per block
}

__device__ __forceinline__ v8f wmma16(v16h a, v16h b, v8f c) {
    // D = A(16x32 f16) * B(32x16 f16) + C(16x16 f32)
    return __builtin_amdgcn_wmma_f32_16x16x32_f16(false, a, false, b, (short)0, c,
                                                  false, false);
}

// 16 contiguous halves from LDS -> one fragment register pair (2x ds_load_b128)
__device__ __forceinline__ v16h lds_frag(const _Float16* p) {
    v16h a;
    ((v8h*)&a)[0] = *(const v8h*)(p);
    ((v8h*)&a)[1] = *(const v8h*)(p + 8);
    return a;
}

__device__ __forceinline__ v8f splat8(float x) {
    v8f v;
#pragma unroll
    for (int i = 0; i < 8; ++i) v[i] = x;
    return v;
}

// hardware-native transcendentals: v_exp_f32 (TRANS, co-executes with XDL) + v_rcp_f32
__device__ __forceinline__ float fsigmoid(float x) {
    return __builtin_amdgcn_rcpf(1.0f + __builtin_amdgcn_exp2f(-1.442695041f * x));
}
__device__ __forceinline__ float ftanh(float x) {
    // tanh(x) = 2*sigmoid(2x) - 1
    return 2.0f * __builtin_amdgcn_rcpf(1.0f + __builtin_amdgcn_exp2f(-2.885390082f * x)) - 1.0f;
}

// cap VGPRs at 256 (4 waves/SIMD): removes s_set_vgpr_msb traffic, raises occupancy
__global__ __launch_bounds__(256, 4)
void mm_decoder_lstm_wmma(const float* __restrict__ traj_rel,
                          const float* __restrict__ h0,
                          const float* __restrict__ c0,
                          const float* __restrict__ W_ih,   // [256,16]
                          const float* __restrict__ W_hh,   // [256,64]
                          const float* __restrict__ b_ih,   // [256]
                          const float* __restrict__ b_hh,   // [256]
                          const float* __restrict__ W_se,   // [16,12]
                          const float* __restrict__ b_se,   // [16]
                          const float* __restrict__ W_pos,  // [12,64]
                          const float* __restrict__ b_pos,  // [12]
                          const float* __restrict__ W_conf, // [6,64]
                          const float* __restrict__ b_conf, // [6]
                          float* __restrict__ pred,         // [B,6,30,2]
                          float* __restrict__ conf_out)     // [B,6]
{
    // ---- LDS: weights pre-swizzled into WMMA B-fragment order ----
    // B-fragment: half j of lane l holds W[N = l%16][K = (j/8)*16 + (l/16)*8 + j%8]
    // Bias-augmentation: sWih/sWse carry the bias in their K==16 row; the
    // matching A fragments carry 1.0 at K==16, so gate/x WMMAs start from C=0.
    __shared__ _Float16 sWhh [16 * 2 * 32 * 16];  // 32KB [tile][kk][lane][j]
    __shared__ _Float16 sWih [16 * 32 * 16];      // 16KB (K 0..15 = W_ih, K16 = bias)
    __shared__ _Float16 sWpos[2 * 32 * 16];       //  2KB (N padded 12->16)
    __shared__ _Float16 sWse [32 * 16];           //  1KB (K 0..11 = W_se, K16 = b_se)
    __shared__ _Float16 sWconf[2 * 32 * 16];      //  2KB (N padded 6->16)
    __shared__ float    sBpos[16];
    __shared__ float    sBconf[16];
    __shared__ _Float16 sH[WAVES][16 * 64];       // per-wave h staging, row-major
    __shared__ _Float16 sX[WAVES][16 * 16];       // per-wave rel/x staging, row-major
    __shared__ float    sConf[WAVES][16 * 8];     // conf logits for softmax

    const int tid = threadIdx.x;

    // ---- prologue: pack fragments (f32 -> f16 convert) ----
    for (int idx = tid; idx < 16 * 2 * 32 * 16; idx += 256) {
        int j = idx & 15, ln = (idx >> 4) & 31, kk = (idx >> 9) & 1, t = idx >> 10;
        int nn = ln & 15, k = kk * 32 + (j >> 3) * 16 + (ln >> 4) * 8 + (j & 7);
        sWhh[idx] = (_Float16)W_hh[(t * 16 + nn) * 64 + k];
    }
    for (int idx = tid; idx < 16 * 32 * 16; idx += 256) {
        int j = idx & 15, ln = (idx >> 4) & 31, t = idx >> 9;
        int nn = ln & 15, k = (j >> 3) * 16 + (ln >> 4) * 8 + (j & 7);
        float v = 0.f;
        if (k < 16)       v = W_ih[(t * 16 + nn) * 16 + k];
        else if (k == 16) v = b_ih[t * 16 + nn] + b_hh[t * 16 + nn];  // bias row
        sWih[idx] = (_Float16)v;
    }
    for (int idx = tid; idx < 2 * 32 * 16; idx += 256) {
        int j = idx & 15, ln = (idx >> 4) & 31, kk = idx >> 9;
        int nn = ln & 15, k = kk * 32 + (j >> 3) * 16 + (ln >> 4) * 8 + (j & 7);
        sWpos[idx] = (nn < 12) ? (_Float16)W_pos[nn * 64 + k] : (_Float16)0.f;
    }
    for (int idx = tid; idx < 32 * 16; idx += 256) {
        int j = idx & 15, ln = idx >> 4;
        int nn = ln & 15, k = (j >> 3) * 16 + (ln >> 4) * 8 + (j & 7);
        float v = 0.f;
        if (k < 12)       v = W_se[nn * 12 + k];
        else if (k == 16) v = b_se[nn];                               // bias row
        sWse[idx] = (_Float16)v;
    }
    for (int idx = tid; idx < 2 * 32 * 16; idx += 256) {
        int j = idx & 15, ln = (idx >> 4) & 31, kk = idx >> 9;
        int nn = ln & 15, k = kk * 32 + (j >> 3) * 16 + (ln >> 4) * 8 + (j & 7);
        sWconf[idx] = (nn < 6) ? (_Float16)W_conf[nn * 64 + k] : (_Float16)0.f;
    }
    if (tid < 16) {
        sBpos[tid]  = (tid < 12) ? b_pos[tid]  : 0.f;
        sBconf[tid] = (tid < 6)  ? b_conf[tid] : 0.f;
    }
    __syncthreads();

    // ---- per-wave setup ----
    const int lane = tid & 31;
    const int wave = tid >> 5;
    const int hi   = lane >> 4;     // lane group (0: lanes 0-15, 1: 16-31)
    const int n    = lane & 15;     // D-layout column / A-layout row
    const int row  = n;
    const int b0   = (blockIdx.x * WAVES + wave) * ROWS;

    _Float16* hst = &sH[wave][0];
    _Float16* xst = &sX[wave][0];

    // h0 -> LDS staging (row-major [16][64] f16): 32 halves per lane
#pragma unroll
    for (int q = 0; q < 32; ++q) {
        int idx = lane * 32 + q;
        hst[idx] = (_Float16)h0[b0 * 64 + idx];
    }
    // c state in registers, D-fragment layout: creg[tile][r] <-> c[r+8*hi][16*tile+n]
    float creg[4][8];
#pragma unroll
    for (int j = 0; j < 4; ++j)
#pragma unroll
        for (int r = 0; r < 8; ++r)
            creg[j][r] = c0[(b0 + r + 8 * hi) * 64 + j * 16 + n];

    // stage rel12 = (x,y) tiled 6 times (x,y,x,y,...) into xst, pad cols 12..15 with 0
    if (hi == 0) {
        float rx = traj_rel[(b0 + n) * 2 + 0];
        float ry = traj_rel[(b0 + n) * 2 + 1];
#pragma unroll
        for (int p = 0; p < 16; ++p)
            xst[n * 16 + p] = (p < 12) ? (_Float16)((p & 1) ? ry : rx) : (_Float16)0.f;
    }

    // A-fragment pad half: zeros except K==16 slot (lanes 0-15, element 0) = 1.0
    // -> multiplies the bias row stored at K==16 of sWih/sWse.
    v8h zpad;
#pragma unroll
    for (int i = 0; i < 8; ++i) zpad[i] = (_Float16)0.f;
    if (hi == 0) zpad[0] = (_Float16)1.0f;

    // loop-invariant B fragments hoisted to registers
    const v16h wse_f   = lds_frag(&sWse[lane * 16]);
    const v16h wpos_f0 = lds_frag(&sWpos[(0 * 32 + lane) * 16]);
    const v16h wpos_f1 = lds_frag(&sWpos[(1 * 32 + lane) * 16]);

    // A-fragment of h from staging: half j of lane l <-> h[l%16][(j/8)*16 + (l/16)*8 + j%8]
    auto hfrag = [&](int half) -> v16h {
        const _Float16* base = &hst[row * 64 + half * 32 + hi * 8];
        v16h a;
        ((v8h*)&a)[0] = *(const v8h*)(base);
        ((v8h*)&a)[1] = *(const v8h*)(base + 16);
        return a;
    };

    v16h ha0, ha1, xa;
    const v8f zc = splat8(0.f);   // folds to inline-0 C operand

    // x = leaky(stage @ W_se^T + b_se): wmma (bias via K16 row), restage, reload
    auto compute_x = [&]() -> v16h {
        v16h ra;  // A-frag of staged rel (K=12 padded; K16 = 1.0 for bias)
        ((v8h*)&ra)[0] = *(const v8h*)&xst[row * 16 + hi * 8];
        ((v8h*)&ra)[1] = zpad;
        v8f acc = wmma16(ra, wse_f, zc);
#pragma unroll
        for (int r = 0; r < 8; ++r) {
            float v = acc[r];
            v = fmaxf(v, 0.01f * v);                       // leaky_relu
            xst[(r + 8 * hi) * 16 + n] = (_Float16)v;      // restage (D -> rows)
        }
        v16h out;
        ((v8h*)&out)[0] = *(const v8h*)&xst[row * 16 + hi * 8];
        ((v8h*)&out)[1] = zpad;                            // K16=1.0 feeds sWih bias row
        return out;
    };

    auto gate_tile = [&](int tile) -> v8f {
        v8f acc = wmma16(xa, lds_frag(&sWih[(tile * 32 + lane) * 16]), zc);  // + bias
        acc = wmma16(ha0, lds_frag(&sWhh[((tile * 2 + 0) * 32 + lane) * 16]), acc);
        acc = wmma16(ha1, lds_frag(&sWhh[((tile * 2 + 1) * 32 + lane) * 16]), acc);
        return acc;
    };

    xa  = compute_x();        // initial x from rel12
    ha0 = hfrag(0);           // initial h fragments (stay live across iterations)
    ha1 = hfrag(1);

#pragma unroll 1
    for (int t = 0; t < T_STEPS; ++t) {
        // gates: 16 N-tiles of 16 cols; order i(0-3) f(4-7) g(8-11) o(12-15).
        // Staggered elementwise keeps at most ~3 accumulators live.
#pragma unroll
        for (int j = 0; j < 4; ++j) {
            v8f gi = gate_tile(j);
            v8f gg = gate_tile(8 + j);
            float ig[8];
#pragma unroll
            for (int r = 0; r < 8; ++r) ig[r] = fsigmoid(gi[r]) * ftanh(gg[r]);
            v8f gf = gate_tile(4 + j);
#pragma unroll
            for (int r = 0; r < 8; ++r) creg[j][r] = fsigmoid(gf[r]) * creg[j][r] + ig[r];
            v8f go = gate_tile(12 + j);
#pragma unroll
            for (int r = 0; r < 8; ++r) {
                float hv = fsigmoid(go[r]) * ftanh(creg[j][r]);
                hst[(r + 8 * hi) * 64 + j * 16 + n] = (_Float16)hv;  // restage h
            }
        }
        // reload h fragments once; they also serve as next iteration's gate operands
        ha0 = hfrag(0);
        ha1 = hfrag(1);
        // rel = h_new @ W_pos^T + b_pos  (K=64 -> 2 WMMAs)
        v8f rel = splat8(sBpos[n]);
        rel = wmma16(ha0, wpos_f0, rel);
        rel = wmma16(ha1, wpos_f1, rel);
#pragma unroll
        for (int r = 0; r < 8; ++r) {
            int brow = b0 + r + 8 * hi;
            if (n < 12)  // pred[b][mode][t][xy]
                pred[((brow * M_MODES + (n >> 1)) * T_STEPS + t) * 2 + (n & 1)] = rel[r];
            xst[(r + 8 * hi) * 16 + n] = (n < 12) ? (_Float16)rel[r] : (_Float16)0.f;
        }
        xa = compute_x();  // x for next step
    }

    // ---- confidences: softmax(h_final @ W_conf^T + b_conf) over 6 modes ----
    v8f cl = splat8(sBconf[n]);
    cl = wmma16(ha0, lds_frag(&sWconf[(0 * 32 + lane) * 16]), cl);
    cl = wmma16(ha1, lds_frag(&sWconf[(1 * 32 + lane) * 16]), cl);
    if (n < 8) {
#pragma unroll
        for (int r = 0; r < 8; ++r) sConf[wave][(r + 8 * hi) * 8 + n] = cl[r];
    }
    if (hi == 0) {  // lanes 0-15: one row each
        float v[6], mx = -1e30f;
#pragma unroll
        for (int k = 0; k < 6; ++k) { v[k] = sConf[wave][n * 8 + k]; mx = fmaxf(mx, v[k]); }
        float s = 0.f;
#pragma unroll
        for (int k = 0; k < 6; ++k) {
            v[k] = __builtin_amdgcn_exp2f(1.442695041f * (v[k] - mx));
            s += v[k];
        }
        float inv = __builtin_amdgcn_rcpf(s);
#pragma unroll
        for (int k = 0; k < 6; ++k) conf_out[(b0 + n) * M_MODES + k] = v[k] * inv;
    }
}

extern "C" void kernel_launch(void* const* d_in, const int* in_sizes, int n_in,
                              void* d_out, int out_size, void* d_ws, size_t ws_size,
                              hipStream_t stream) {
    // setup_inputs order: traj_abs, traj_rel, h0, c0, W_ih, W_hh, b_ih, b_hh,
    //                     W_se, b_se, W_pos, b_pos, W_conf, b_conf   (all f32)
    const float* traj_rel = (const float*)d_in[1];
    const float* h0       = (const float*)d_in[2];
    const float* c0       = (const float*)d_in[3];
    const float* W_ih     = (const float*)d_in[4];
    const float* W_hh     = (const float*)d_in[5];
    const float* b_ih     = (const float*)d_in[6];
    const float* b_hh     = (const float*)d_in[7];
    const float* W_se     = (const float*)d_in[8];
    const float* b_se     = (const float*)d_in[9];
    const float* W_pos    = (const float*)d_in[10];
    const float* b_pos    = (const float*)d_in[11];
    const float* W_conf   = (const float*)d_in[12];
    const float* b_conf   = (const float*)d_in[13];

    const int btot = in_sizes[2] / H_DIM;          // B from h0 element count
    float* pred = (float*)d_out;                   // [B,6,30,2]
    float* conf = pred + (size_t)btot * M_MODES * T_STEPS * 2;  // [B,6]

    dim3 grid(btot / BLK_B), block(256);
    mm_decoder_lstm_wmma<<<grid, block, 0, stream>>>(
        traj_rel, h0, c0, W_ih, W_hh, b_ih, b_hh,
        W_se, b_se, W_pos, b_pos, W_conf, b_conf, pred, conf);
}
